// tModLodaer_t_19593640804842
// MI455X (gfx1250) — matrix operand order repaired and checked by reading, not compiled, not run on hardware
//
#include <hip/hip_runtime.h>
#include <hip/hip_bf16.h>
#include <math.h>

// ---------------------------------------------------------------------------
// CDNA5 (gfx1250) WMMA types
// ---------------------------------------------------------------------------
typedef __attribute__((ext_vector_type(16))) _Float16 v16h;
typedef __attribute__((ext_vector_type(8)))  float    v8f;

#define FEAT 64
#define NHEAD 2
#define HDIM 32

// Async global->LDS staging (gfx1250 GLOBAL_LOAD_ASYNC_TO_LDS_B128, ASYNCcnt).
// Guarded: falls back to direct global loads if the builtin is unavailable.
#if defined(__HIP_DEVICE_COMPILE__) && __has_builtin(__builtin_amdgcn_global_load_async_to_lds_b128)
#define USE_ASYNC_LDS 1
#else
#define USE_ASYNC_LDS 0
#endif

#if USE_ASYNC_LDS
typedef __attribute__((vector_size(16))) int vi4;   // builtin expects vi4* args
__device__ __forceinline__ void wait_async0() {
#if __has_builtin(__builtin_amdgcn_s_wait_asynccnt)
  __builtin_amdgcn_s_wait_asynccnt(0);
#else
  asm volatile("s_wait_asynccnt 0" ::: "memory");
#endif
}
#endif

// Up to 5 concatenated 64-wide segments form the logical A matrix of a GEMM.
// idx[s] == nullptr means "direct row", else row r reads base[s][idx[s][r]*64 + off].
struct SegDesc {
  const float* base[5];
  const int*   idx[5];
};

template<int ACT>
__device__ __forceinline__ float actf(float x) {
  if (ACT == 1) return x / (1.f + __expf(-x));                       // silu
  if (ACT == 2) return 1.f / (1.f + __expf(-x));                     // sigmoid
  if (ACT == 3) return 0.5f * x * (1.f + erff(x * 0.70710678118f));  // gelu (exact)
  return x;
}

// ---------------------------------------------------------------------------
// Fused gather-concat GEMM:  C[rows x Nout] = act(A_gathered @ W + bias)
//   - A fp32: NSEG==0 -> dense [rows x (nkt*32)] (lda, K multiple of 32)
//             NSEG>0  -> NSEG gathered segments of width 64 (K = NSEG*64)
//   - Wp: f16, pre-packed in exact B-fragment order:
//         Wp[(((colTile16 * nkt) + kt) * 32 + lane) * 16 + j]
//     The packed tile for this block's colTile is staged once into LDS via
//     async global->LDS copies and shared by all 4 waves (4x B reuse).
//   - one 16x16 C tile per wave, 4 waves per block, K-step 32 via
//     v_wmma_f32_16x16x32_f16 (f32 accumulate)
// Per K-step: 4 unconditional b128 A loads + 32B LDS B load + 1 WMMA.
// VGPR layouts per CDNA5 ISA 7.12.2.
// ---------------------------------------------------------------------------
template<int NSEG, int ACT>
__global__ __launch_bounds__(128) void wmma_gemm_kernel(
    const float* __restrict__ Adense, int lda, int nkt_rt,
    SegDesc segs,
    const _Float16* __restrict__ Wp, const float* __restrict__ bias,
    float* __restrict__ C, int ldc, int rows)
{
  const int wave = threadIdx.x >> 5;
  const int lane = threadIdx.x & 31;
  const int half = lane >> 4;
  const int lr   = lane & 15;
  const int rowTile = blockIdx.x * 64 + wave * 16;
  const int colTile = blockIdx.y * 16;

  const int nkt = (NSEG > 0) ? (NSEG * 2) : nkt_rt;

#if USE_ASYNC_LDS
  // Stage this block's packed weight tile (nkt KB) into LDS, all threads help.
  __shared__ _Float16 wlds[16 * 32 * 16];   // supports nkt <= 16 (16 KB)
  {
    const _Float16* wsrc = Wp + (size_t)blockIdx.y * nkt * (32 * 16);
    for (int j = threadIdx.x; j < nkt * 64; j += 128) {   // 16-byte chunks
      __builtin_amdgcn_global_load_async_to_lds_b128(
          (vi4*)(wsrc + (size_t)j * 8),
          (vi4*)(wlds + (size_t)j * 8),
          0, 0);
    }
    wait_async0();
    __syncthreads();
  }
  const _Float16* wptr = wlds + (size_t)lane * 16;
#else
  const _Float16* wptr = Wp + (((size_t)blockIdx.y * nkt) * 32 + lane) * 16;
#endif

  if (rowTile >= rows) return;   // uniform per-wave (after barrier): EXEC all-ones for WMMA

  v8f acc;
  const float bv = bias[colTile + lr];
#pragma unroll
  for (int v = 0; v < 8; ++v) acc[v] = bv;

  const int m  = rowTile + lr;
  const int am = (m < rows) ? m : (rows - 1);   // clamp (no EXEC change)

  // Hoist gather row-base pointers out of the K loop (one index load per segment).
  const float* rowbase[(NSEG > 0) ? NSEG : 1];
  if (NSEG > 0) {
#pragma unroll
    for (int s = 0; s < ((NSEG > 0) ? NSEG : 1); ++s) {
      const int* ip = segs.idx[s];
      const int r = ip ? ip[am] : am;
      rowbase[s] = segs.base[s] + (size_t)r * 64;
    }
  }

  auto step = [&](int kt) {
    const float* rp;
    if (NSEG > 0) rp = rowbase[kt >> 1] + ((kt & 1) << 5) + (half << 3);
    else          rp = Adense + (size_t)am * lda + (kt << 5) + (half << 3);
    const float4 a0 = *(const float4*)(rp);
    const float4 a1 = *(const float4*)(rp + 4);
    const float4 a2 = *(const float4*)(rp + 16);
    const float4 a3 = *(const float4*)(rp + 20);
    v16h afrag;
    afrag[0]  = (_Float16)a0.x;  afrag[1]  = (_Float16)a0.y;
    afrag[2]  = (_Float16)a0.z;  afrag[3]  = (_Float16)a0.w;
    afrag[4]  = (_Float16)a1.x;  afrag[5]  = (_Float16)a1.y;
    afrag[6]  = (_Float16)a1.z;  afrag[7]  = (_Float16)a1.w;
    afrag[8]  = (_Float16)a2.x;  afrag[9]  = (_Float16)a2.y;
    afrag[10] = (_Float16)a2.z;  afrag[11] = (_Float16)a2.w;
    afrag[12] = (_Float16)a3.x;  afrag[13] = (_Float16)a3.y;
    afrag[14] = (_Float16)a3.z;  afrag[15] = (_Float16)a3.w;
    const v16h bfrag = *(const v16h*)(wptr + (size_t)kt * (32 * 16));
    acc = __builtin_amdgcn_wmma_f32_16x16x32_f16(false, afrag, false, bfrag,
                                                 (short)0, acc, false, false);
  };

  if (NSEG > 0) {
#pragma unroll
    for (int kt = 0; kt < NSEG * 2; ++kt) step(kt);
  } else {
    for (int kt = 0; kt < nkt; ++kt) step(kt);
  }

#pragma unroll
  for (int v = 0; v < 8; ++v) {
    const int r = rowTile + v + half * 8;
    if (r < rows) C[(size_t)r * ldc + colTile + lr] = actf<ACT>(acc[v]);
  }
}

// ---------------------------------------------------------------------------
// Weight pre-pack: fp32 [K x Nout] -> f16 in B-fragment order, zero-padded
// to nkt*32 K-rows.  One dword (2 halfs) per thread.
// ---------------------------------------------------------------------------
__global__ void k_packW(const float* __restrict__ W, _Float16* __restrict__ P,
                        int K, int Nout, int nkt)
{
  const int i = blockIdx.x * blockDim.x + threadIdx.x;
  const int total = (Nout / 16) * nkt * 32 * 8;
  if (i >= total) return;
  const int v    = i & 7;
  const int lane = (i >> 3) & 31;
  const int kt   = (i >> 8) % nkt;
  const int ct   = (i >> 8) / nkt;
  const int half = lane >> 4, lr = lane & 15;
  const int k    = kt * 32 + (v >> 2) * 16 + half * 8 + (v & 3) * 2;
  const int col  = ct * 16 + lr;
  const float x0 = (k     < K) ? W[(size_t)k * Nout + col]       : 0.f;
  const float x1 = (k + 1 < K) ? W[(size_t)(k + 1) * Nout + col] : 0.f;
  _Float16* dst = P + (((size_t)ct * nkt + kt) * 32 + lane) * 16 + v * 2;
  dst[0] = (_Float16)x0;
  dst[1] = (_Float16)x1;
}

// ---------------------------------------------------------------------------
// Small support kernels
// ---------------------------------------------------------------------------
__global__ void k_embed(const int* __restrict__ idx, const float* __restrict__ emb,
                        float* __restrict__ af)
{
  const int r = blockIdx.x, t = threadIdx.x;   // 64 threads / row
  __shared__ float red[64];
  const float v = emb[(size_t)idx[r] * 64 + t];
  red[t] = v * v;  __syncthreads();
  for (int off = 32; off > 0; off >>= 1) { if (t < off) red[t] += red[t + off]; __syncthreads(); }
  const float nrm = sqrtf(red[0]);
  const float sc  = (nrm > 1.f) ? (1.f / (nrm + 1e-7f)) : 1.f;
  af[(size_t)r * 64 + t] = v * sc;
}

// 16 radial-basis features padded to a 32-wide row (cols 16..31 = 0) so the
// downstream WMMA GEMM runs a guard-free K=32.
__global__ void k_ebf32(const float* __restrict__ d, float* __restrict__ out,
                        int n, float rcut)
{
  const int i = blockIdx.x * blockDim.x + threadIdx.x;
  if (i >= n * 32) return;
  const int r = i >> 5, f = i & 31;
  const float dv = d[r];
  out[i] = (f < 16)
      ? sqrtf(2.f / rcut) * __sinf((float)f * 3.14159265358979f / rcut * dv) / dv
      : 0.f;
}

__global__ void k_prep_edges(const int* __restrict__ nbr, int* __restrict__ srca,
                             int* __restrict__ dsta, int M)
{
  const int m = blockIdx.x * blockDim.x + threadIdx.x;
  if (m >= M) return;
  srca[m] = nbr[2 * m];
  dsta[m] = nbr[2 * m + 1];
}

__global__ void k_prep_triples(const int* __restrict__ bpi, const int* __restrict__ srca,
                               const int* __restrict__ dsta, int* __restrict__ bp0a,
                               int* __restrict__ bp1a, int* __restrict__ tia,
                               int* __restrict__ tja, int* __restrict__ tka, int L)
{
  const int l = blockIdx.x * blockDim.x + threadIdx.x;
  if (l >= L) return;
  const int b0 = bpi[2 * l], b1 = bpi[2 * l + 1];
  bp0a[l] = b0; bp1a[l] = b1;
  tia[l] = srca[b1]; tja[l] = dsta[b0]; tka[l] = dsta[b1];
}

__global__ void k_mul2(float* __restrict__ out, const float* __restrict__ a,
                       const float* __restrict__ b, int n)
{
  const int i = blockIdx.x * blockDim.x + threadIdx.x;
  if (i < n) out[i] = a[i] * b[i];
}

__global__ void k_mulacc(float* __restrict__ out, const float* __restrict__ a,
                         const float* __restrict__ b, int n)  // out += a*b
{
  const int i = blockIdx.x * blockDim.x + threadIdx.x;
  if (i < n) out[i] += a[i] * b[i];
}

__global__ void k_add_inplace(float* __restrict__ a, const float* __restrict__ b, int n)
{
  const int i = blockIdx.x * blockDim.x + threadIdx.x;
  if (i < n) a[i] += b[i];
}

__global__ void k_copy(float* __restrict__ a, const float* __restrict__ b, int n)
{
  const int i = blockIdx.x * blockDim.x + threadIdx.x;
  if (i < n) a[i] = b[i];
}

// triple gate: e[tgt] += sig(fea@w1) * silu(fea@w2) * bj * bk * (a*angw + angb)
__global__ void k_tri_scatter(float* __restrict__ e, const float* __restrict__ t1,
                              const float* __restrict__ t2, const float* __restrict__ bj,
                              const float* __restrict__ bk, const float* __restrict__ ajik,
                              const int* __restrict__ tgt, const float* __restrict__ angw,
                              const float* __restrict__ angb, int rows)
{
  const int i = blockIdx.x * blockDim.x + threadIdx.x;
  if (i >= rows * 64) return;
  const int l = i >> 6, c = i & 63;
  const float ang = ajik[l] * angw[c] + angb[c];
  atomicAdd(&e[(size_t)tgt[l] * 64 + c], t1[i] * t2[i] * bj[i] * bk[i] * ang);
}

// conv scatter: af[src] += silu(gt@w1)*sig(gt@w2)*(bd@wr)
__global__ void k_cv_scatter(float* __restrict__ af, const float* __restrict__ c1,
                             const float* __restrict__ c2, const float* __restrict__ cwr,
                             const int* __restrict__ srca, int M)
{
  const int i = blockIdx.x * blockDim.x + threadIdx.x;
  if (i >= M * 64) return;
  const int m = i >> 6, c = i & 63;
  atomicAdd(&af[(size_t)srca[m] * 64 + c], c1[i] * c2[i] * cwr[i]);
}

__global__ void k_ln(float* __restrict__ out, const float* __restrict__ in,
                     const float* __restrict__ res, const float* __restrict__ s,
                     const float* __restrict__ bb)
{
  const int r = blockIdx.x, t = threadIdx.x;   // 64 threads / row
  __shared__ float red[64];
  float x = in[(size_t)r * 64 + t];
  if (res) x += res[(size_t)r * 64 + t];
  red[t] = x;  __syncthreads();
  for (int off = 32; off > 0; off >>= 1) { if (t < off) red[t] += red[t + off]; __syncthreads(); }
  const float mean = red[0] * (1.f / 64.f);
  __syncthreads();
  const float d = x - mean;
  red[t] = d * d;  __syncthreads();
  for (int off = 32; off > 0; off >>= 1) { if (t < off) red[t] += red[t + off]; __syncthreads(); }
  const float var = red[0] * (1.f / 64.f);
  out[(size_t)r * 64 + t] = d * rsqrtf(var + 1e-5f) * s[t] + bb[t];
}

// masked softmax attention, one (graph, head) per block, 64 threads (one query each)
__global__ void k_attn(const float* __restrict__ qkv, const int* __restrict__ n_atoms,
                       float* __restrict__ o, int npg)
{
  const int b = blockIdx.x, h = blockIdx.y, i = threadIdx.x;
  __shared__ float kb[64 * HDIM];
  __shared__ float vb[64 * HDIM];
  const float* base = qkv + (size_t)b * npg * (3 * FEAT);
  for (int t = i; t < npg * HDIM; t += npg) {
    const int j = t / HDIM, d = t % HDIM;
    kb[t] = base[(size_t)j * (3 * FEAT) + FEAT     + h * HDIM + d];
    vb[t] = base[(size_t)j * (3 * FEAT) + 2 * FEAT + h * HDIM + d];
  }
  __syncthreads();
  float q[HDIM];
#pragma unroll
  for (int d = 0; d < HDIM; ++d) q[d] = base[(size_t)i * (3 * FEAT) + h * HDIM + d];
  const int na = n_atoms[b];
  const float scale = 0.17677669529663687f;  // 1/sqrt(32)
  float mmax = -1e30f;
  for (int j = 0; j < npg; ++j) {
    float sc = 0.f;
    for (int d = 0; d < HDIM; ++d) sc += q[d] * kb[j * HDIM + d];
    sc = sc * scale + ((j >= na) ? -1e9f : 0.f);
    mmax = fmaxf(mmax, sc);
  }
  float acc[HDIM];
#pragma unroll
  for (int d = 0; d < HDIM; ++d) acc[d] = 0.f;
  float ssum = 0.f;
  for (int j = 0; j < npg; ++j) {
    float sc = 0.f;
    for (int d = 0; d < HDIM; ++d) sc += q[d] * kb[j * HDIM + d];
    sc = sc * scale + ((j >= na) ? -1e9f : 0.f);
    const float p = __expf(sc - mmax);
    ssum += p;
    for (int d = 0; d < HDIM; ++d) acc[d] += p * vb[j * HDIM + d];
  }
  const float inv = 1.f / ssum;
  for (int d = 0; d < HDIM; ++d)
    o[((size_t)b * npg + i) * FEAT + h * HDIM + d] = acc[d] * inv;
}

__global__ void k_segsum(float* __restrict__ cry, const float* __restrict__ af,
                         const int* __restrict__ n_atoms, int npg)
{
  const int b = blockIdx.x, c = threadIdx.x;   // 64 threads
  const int na = n_atoms[b];
  float s = 0.f;
  for (int i = 0; i < na && i < npg; ++i) s += af[((size_t)b * npg + i) * 64 + c];
  cry[(size_t)b * 64 + c] = s;
}

__global__ void k_out(float* __restrict__ out, const float* __restrict__ cry,
                      const float* __restrict__ w, const float* __restrict__ b, int B)
{
  const int i = blockIdx.x * blockDim.x + threadIdx.x;
  if (i >= B) return;
  float s = 0.f;
  for (int c = 0; c < 64; ++c) s += cry[(size_t)i * 64 + c] * w[c];
  out[i] = s + b[0];
}

// ---------------------------------------------------------------------------
// Host-side GEMM dispatch
// ---------------------------------------------------------------------------
static void launch_gemm(hipStream_t st, int nseg, int act,
                        const float* A, int lda, int nkt, const SegDesc& segs,
                        const _Float16* Wp, const float* bias, float* C, int ldc,
                        int rows, int Nout)
{
  dim3 grid((rows + 63) / 64, Nout / 16);
  dim3 block(128);
#define GO(NS, AC) wmma_gemm_kernel<NS, AC><<<grid, block, 0, st>>>(A, lda, nkt, segs, Wp, bias, C, ldc, rows)
  switch (nseg) {
    case 0:
      switch (act) { case 0: GO(0,0); break; case 1: GO(0,1); break;
                     case 2: GO(0,2); break; default: GO(0,3); break; } break;
    case 3:
      switch (act) { case 0: GO(3,0); break; case 1: GO(3,1); break;
                     case 2: GO(3,2); break; default: GO(3,3); break; } break;
    default:
      switch (act) { case 0: GO(5,0); break; case 1: GO(5,1); break;
                     case 2: GO(5,2); break; default: GO(5,3); break; } break;
  }
#undef GO
}

static inline dim3 g1d(long n, int bs) { return dim3((unsigned)((n + bs - 1) / bs)); }

// ---------------------------------------------------------------------------
extern "C" void kernel_launch(void* const* d_in, const int* in_sizes, int n_in,
                              void* d_out, int out_size, void* d_ws, size_t ws_size,
                              hipStream_t stream)
{
  (void)n_in; (void)out_size; (void)ws_size;
  // ---- inputs (setup_inputs dict order, params flattened in insertion order) ----
  const int*   atom_idx = (const int*)  d_in[0];
  const float* bonds_r  = (const float*)d_in[1];
  const int*   n_atoms  = (const int*)  d_in[2];
  const float* t_ij     = (const float*)d_in[3];
  const float* t_ik     = (const float*)d_in[4];
  const float* a_jik    = (const float*)d_in[5];
  const int*   nbr      = (const int*)  d_in[6];
  const int*   bpi      = (const int*)  d_in[8];   // [L,2]; col1 == tgt
  const float* emb      = (const float*)d_in[9];
  const float* wb_w   = (const float*)d_in[10]; const float* wb_b   = (const float*)d_in[11];
  const float* weij_w = (const float*)d_in[12]; const float* weij_b = (const float*)d_in[13];
  const float* wr_w   = (const float*)d_in[14]; const float* wr_b   = (const float*)d_in[15];
  const float* fc1_w  = (const float*)d_in[16]; const float* fc1_b  = (const float*)d_in[17];
  const float* fc2_w  = (const float*)d_in[18]; const float* fc2_b  = (const float*)d_in[19];
  const float* out_w  = (const float*)d_in[20]; const float* out_b  = (const float*)d_in[21];
  const float* tb_ang_w = (const float*)d_in[22]; const float* tb_ang_b = (const float*)d_in[23];
  const float* tb_bk_w  = (const float*)d_in[24]; const float* tb_bk_b  = (const float*)d_in[25];
  const float* tb_bj_w  = (const float*)d_in[26]; const float* tb_bj_b  = (const float*)d_in[27];
  const float* tb_fea_w = (const float*)d_in[28]; const float* tb_fea_b = (const float*)d_in[29];
  const float* tb_w1_w  = (const float*)d_in[30]; const float* tb_w1_b  = (const float*)d_in[31];
  const float* tb_w2_w  = (const float*)d_in[32]; const float* tb_w2_b  = (const float*)d_in[33];
  const float* eu_w1_w  = (const float*)d_in[34]; const float* eu_w1_b  = (const float*)d_in[35];
  const float* eu_w2_w  = (const float*)d_in[36]; const float* eu_w2_b  = (const float*)d_in[37];
  const float* eu_wr_w  = (const float*)d_in[38]; const float* eu_wr_b  = (const float*)d_in[39];
  const float* eu_w3_w  = (const float*)d_in[40]; const float* eu_w3_b  = (const float*)d_in[41];
  const float* cv_full_w= (const float*)d_in[42]; const float* cv_full_b= (const float*)d_in[43];
  const float* cv_wr_w  = (const float*)d_in[44]; const float* cv_wr_b  = (const float*)d_in[45];
  const float* cv_w1_w  = (const float*)d_in[46]; const float* cv_w1_b  = (const float*)d_in[47];
  const float* cv_w2_w  = (const float*)d_in[48]; const float* cv_w2_b  = (const float*)d_in[49];
  const float* tf_in_w  = (const float*)d_in[50]; const float* tf_in_b  = (const float*)d_in[51];
  const float* tf_out_w = (const float*)d_in[52]; const float* tf_out_b = (const float*)d_in[53];
  const float* tf_l1_w  = (const float*)d_in[54]; const float* tf_l1_b  = (const float*)d_in[55];
  const float* tf_l2_w  = (const float*)d_in[56]; const float* tf_l2_b  = (const float*)d_in[57];
  const float* ln1_s = (const float*)d_in[58]; const float* ln1_b = (const float*)d_in[59];
  const float* ln2_s = (const float*)d_in[60]; const float* ln2_b = (const float*)d_in[61];
  const float* n1_s  = (const float*)d_in[62]; const float* n1_b  = (const float*)d_in[63];
  const float* n2_s  = (const float*)d_in[64]; const float* n2_b  = (const float*)d_in[65];

  const int N  = in_sizes[0];          // atoms
  const int M  = in_sizes[1];          // bonds
  const int Bn = in_sizes[2];          // graphs
  const int L  = in_sizes[3];          // triples
  const int nl = in_sizes[22] / 64;    // layers (tb_ang_w is NL*1*64)
  const int NPG = N / Bn;
  const int MAXL = 8;                  // max layers supported for pack tables

  // ---- workspace carve-up (256B aligned) ----
  char* wp = (char*)d_ws;
  auto allocF = [&](size_t n) { float* p = (float*)wp; wp += ((n * 4 + 255) & ~(size_t)255); return p; };
  auto allocI = [&](size_t n) { int*   p = (int*)wp;   wp += ((n * 4 + 255) & ~(size_t)255); return p; };
  auto allocH = [&](size_t n) { _Float16* p = (_Float16*)wp; wp += ((n * 2 + 255) & ~(size_t)255); return p; };

  float* af   = allocF((size_t)N * 64);
  float* xb   = allocF((size_t)N * 64);
  float* ebuf = allocF((size_t)M * 64);
  float* bd   = allocF((size_t)M * 32);      // 16 basis cols + 16 zero-pad
  float* tijb = allocF((size_t)L * 32);
  float* tikb = allocF((size_t)L * 32);
  int* srca = allocI(M);  int* dsta = allocI(M);
  int* bp0a = allocI(L);  int* bp1a = allocI(L);
  int* tia  = allocI(L);  int* tja  = allocI(L);  int* tka = allocI(L);
  float* tA = allocF((size_t)M * 64);
  float* tB = allocF((size_t)M * 64);
  float* tC = allocF((size_t)M * 64);
  float* tD = allocF((size_t)M * 64);
  float* tE = allocF((size_t)M * 64);
  float* tG = allocF((size_t)M * 128);
  float* qkvb = allocF((size_t)N * 192);
  float* ffb  = allocF((size_t)N * 256);
  float* cry  = allocF((size_t)Bn * 64);
  float* cry2 = allocF((size_t)Bn * 64);

  // ---- pre-pack every weight matrix to f16 B-fragment order ----
  auto packW = [&](const float* W, int K, int Nout) -> const _Float16* {
    const int nkt = (K + 31) / 32;
    const size_t n = (size_t)(Nout / 16) * nkt * 32 * 16;
    _Float16* p = allocH(n);
    k_packW<<<g1d((long)(n / 2), 256), 256, 0, stream>>>(W, p, K, Nout, nkt);
    return p;
  };

  const _Float16* P_wb   = packW(wb_w,   16, 64);
  const _Float16* P_weij = packW(weij_w, 192, 64);
  const _Float16* P_wr   = packW(wr_w,   16, 64);
  const _Float16* P_fc1  = packW(fc1_w,  64, 64);
  const _Float16* P_fc2  = packW(fc2_w,  64, 64);

  const _Float16 *P_fea[MAXL], *P_w1[MAXL], *P_w2[MAXL], *P_bj[MAXL], *P_bk[MAXL];
  const _Float16 *P_eu1[MAXL], *P_eu2[MAXL], *P_euwr[MAXL], *P_eu3[MAXL];
  const _Float16 *P_cvf[MAXL], *P_cvwr[MAXL], *P_cv1[MAXL], *P_cv2[MAXL];
  const _Float16 *P_tin[MAXL], *P_tout[MAXL], *P_tl1[MAXL], *P_tl2[MAXL];
  for (int li = 0; li < nl && li < MAXL; ++li) {
    P_fea[li]  = packW(tb_fea_w + (size_t)li * 320 * 64, 320, 64);
    P_w1[li]   = packW(tb_w1_w  + (size_t)li * 64 * 64,  64,  64);
    P_w2[li]   = packW(tb_w2_w  + (size_t)li * 64 * 64,  64,  64);
    P_bj[li]   = packW(tb_bj_w  + (size_t)li * 16 * 64,  16,  64);
    P_bk[li]   = packW(tb_bk_w  + (size_t)li * 16 * 64,  16,  64);
    P_eu1[li]  = packW(eu_w1_w  + (size_t)li * 192 * 64, 192, 64);
    P_eu2[li]  = packW(eu_w2_w  + (size_t)li * 192 * 64, 192, 64);
    P_euwr[li] = packW(eu_wr_w  + (size_t)li * 16 * 64,  16,  64);
    P_eu3[li]  = packW(eu_w3_w  + (size_t)li * 64 * 64,  64,  64);
    P_cvf[li]  = packW(cv_full_w+ (size_t)li * 192 * 128,192, 128);
    P_cvwr[li] = packW(cv_wr_w  + (size_t)li * 16 * 64,  16,  64);
    P_cv1[li]  = packW(cv_w1_w  + (size_t)li * 128 * 64, 128, 64);
    P_cv2[li]  = packW(cv_w2_w  + (size_t)li * 128 * 64, 128, 64);
    P_tin[li]  = packW(tf_in_w  + (size_t)li * 64 * 192, 64,  192);
    P_tout[li] = packW(tf_out_w + (size_t)li * 64 * 64,  64,  64);
    P_tl1[li]  = packW(tf_l1_w  + (size_t)li * 64 * 256, 64,  256);
    P_tl2[li]  = packW(tf_l2_w  + (size_t)li * 256 * 64, 256, 64);
  }

  SegDesc none = {};

  // ---- stage 0: indices, embedding, basis expansion, initial edge features ----
  k_prep_edges<<<g1d(M, 256), 256, 0, stream>>>(nbr, srca, dsta, M);
  k_prep_triples<<<g1d(L, 256), 256, 0, stream>>>(bpi, srca, dsta, bp0a, bp1a, tia, tja, tka, L);
  k_embed<<<dim3(N), dim3(64), 0, stream>>>(atom_idx, emb, af);
  k_ebf32<<<g1d((long)M * 32, 256), 256, 0, stream>>>(bonds_r, bd, M, 5.0f);
  k_ebf32<<<g1d((long)L * 32, 256), 256, 0, stream>>>(t_ij, tijb, L, 3.5f);
  k_ebf32<<<g1d((long)L * 32, 256), 256, 0, stream>>>(t_ik, tikb, L, 3.5f);

  // e0 = bd @ wb + b
  launch_gemm(stream, 0, 0, bd, 32, 1, none, P_wb, wb_b, tB, 64, M, 64);
  // e = (cat(af[src], af[dst], e0) @ weij + b) * (bd @ wr + b)
  {
    SegDesc sd = {{af, af, tB, nullptr, nullptr}, {srca, dsta, nullptr, nullptr, nullptr}};
    launch_gemm(stream, 3, 0, nullptr, 0, 6, sd, P_weij, weij_b, tC, 64, M, 64);
  }
  launch_gemm(stream, 0, 0, bd, 32, 1, none, P_wr, wr_b, tD, 64, M, 64);
  k_mul2<<<g1d((long)M * 64, 256), 256, 0, stream>>>(ebuf, tC, tD, M * 64);

  const int CH = M;                      // triple chunk size (bounds workspace)
  const int nchunks = (L + CH - 1) / CH;

  for (int li = 0; li < nl && li < MAXL; ++li) {
    // ---------------- triple-bond block (chunked over L) ----------------
    for (int c = 0; c < nchunks; ++c) {
      const int r0 = c * CH;
      const int rows = (L - r0 < CH) ? (L - r0) : CH;
      SegDesc sd = {{af, af, af, ebuf, ebuf},
                    {tia + r0, tja + r0, tka + r0, bp0a + r0, bp1a + r0}};
      launch_gemm(stream, 5, 1, nullptr, 0, 10, sd, P_fea[li],
                  tb_fea_b + (size_t)li * 64, tA, 64, rows, 64);
      launch_gemm(stream, 0, 2, tA, 64, 2, none, P_w1[li],
                  tb_w1_b + (size_t)li * 64, tB, 64, rows, 64);
      launch_gemm(stream, 0, 1, tA, 64, 2, none, P_w2[li],
                  tb_w2_b + (size_t)li * 64, tC, 64, rows, 64);
      launch_gemm(stream, 0, 0, tijb + (size_t)r0 * 32, 32, 1, none, P_bj[li],
                  tb_bj_b + (size_t)li * 64, tD, 64, rows, 64);
      launch_gemm(stream, 0, 0, tikb + (size_t)r0 * 32, 32, 1, none, P_bk[li],
                  tb_bk_b + (size_t)li * 64, tE, 64, rows, 64);
      k_tri_scatter<<<g1d((long)rows * 64, 256), 256, 0, stream>>>(
          ebuf, tB, tC, tD, tE, a_jik + r0, bp1a + r0,
          tb_ang_w + (size_t)li * 64, tb_ang_b + (size_t)li * 64, rows);
    }

    // ---------------- edge update ----------------
    {
      SegDesc sd = {{af, af, ebuf, nullptr, nullptr}, {srca, dsta, nullptr, nullptr, nullptr}};
      launch_gemm(stream, 3, 1, nullptr, 0, 6, sd, P_eu1[li],
                  eu_w1_b + (size_t)li * 64, tB, 64, M, 64);
      launch_gemm(stream, 3, 2, nullptr, 0, 6, sd, P_eu2[li],
                  eu_w2_b + (size_t)li * 64, tC, 64, M, 64);
    }
    k_mul2<<<g1d((long)M * 64, 256), 256, 0, stream>>>(tB, tB, tC, M * 64);
    launch_gemm(stream, 0, 1, tB, 64, 2, none, P_eu3[li],
                eu_w3_b + (size_t)li * 64, tC, 64, M, 64);
    launch_gemm(stream, 0, 0, bd, 32, 1, none, P_euwr[li],
                eu_wr_b + (size_t)li * 64, tD, 64, M, 64);
    k_mulacc<<<g1d((long)M * 64, 256), 256, 0, stream>>>(ebuf, tC, tD, M * 64);

    // ---------------- graph conv -> atom features ----------------
    {
      SegDesc sd = {{af, af, ebuf, nullptr, nullptr}, {srca, dsta, nullptr, nullptr, nullptr}};
      launch_gemm(stream, 3, 1, nullptr, 0, 6, sd, P_cvf[li],
                  cv_full_b + (size_t)li * 128, tG, 128, M, 128);
    }
    launch_gemm(stream, 0, 1, tG, 128, 4, none, P_cv1[li],
                cv_w1_b + (size_t)li * 64, tB, 64, M, 64);
    launch_gemm(stream, 0, 2, tG, 128, 4, none, P_cv2[li],
                cv_w2_b + (size_t)li * 64, tC, 64, M, 64);
    launch_gemm(stream, 0, 0, bd, 32, 1, none, P_cvwr[li],
                cv_wr_b + (size_t)li * 64, tD, 64, M, 64);
    k_cv_scatter<<<g1d((long)M * 64, 256), 256, 0, stream>>>(af, tB, tC, tD, srca, M);
    k_ln<<<dim3(N), dim3(64), 0, stream>>>(af, af, nullptr,
                                           n1_s + (size_t)li * 64, n1_b + (size_t)li * 64);
    k_copy<<<g1d((long)N * 64, 256), 256, 0, stream>>>(xb, af, N * 64);

    // ---------------- per-graph transformer ----------------
    launch_gemm(stream, 0, 0, xb, 64, 2, none, P_tin[li],
                tf_in_b + (size_t)li * 192, qkvb, 192, N, 192);
    k_attn<<<dim3(Bn, NHEAD), dim3(NPG), 0, stream>>>(qkvb, n_atoms, tB, NPG);
    launch_gemm(stream, 0, 0, tB, 64, 2, none, P_tout[li],
                tf_out_b + (size_t)li * 64, tC, 64, N, 64);
    k_ln<<<dim3(N), dim3(64), 0, stream>>>(xb, xb, tC,
                                           ln1_s + (size_t)li * 64, ln1_b + (size_t)li * 64);
    launch_gemm(stream, 0, 3, xb, 64, 2, none, P_tl1[li],
                tf_l1_b + (size_t)li * 256, ffb, 256, N, 256);
    launch_gemm(stream, 0, 0, ffb, 256, 8, none, P_tl2[li],
                tf_l2_b + (size_t)li * 64, tC, 64, N, 64);
    k_ln<<<dim3(N), dim3(64), 0, stream>>>(tB, xb, tC,
                                           ln2_s + (size_t)li * 64, ln2_b + (size_t)li * 64);
    k_ln<<<dim3(N), dim3(64), 0, stream>>>(tC, tB, nullptr,
                                           n2_s + (size_t)li * 64, n2_b + (size_t)li * 64);
    k_add_inplace<<<g1d((long)N * 64, 256), 256, 0, stream>>>(af, tC, N * 64);
  }

  // ---------------- readout ----------------
  k_segsum<<<dim3(Bn), dim3(64), 0, stream>>>(cry, af, n_atoms, NPG);
  launch_gemm(stream, 0, 1, cry, 64, 2, none, P_fc1, fc1_b, cry2, 64, Bn, 64);
  launch_gemm(stream, 0, 1, cry2, 64, 2, none, P_fc2, fc2_b, cry, 64, Bn, 64);
  k_out<<<g1d(Bn, 64), 64, 0, stream>>>((float*)d_out, cry, out_w, out_b, Bn);
}
